// ZambaBlock_40261023433412
// MI455X (gfx1250) — matrix operand order repaired and checked by reading, not compiled, not run on hardware
//
#include <hip/hip_runtime.h>
#include <hip/hip_bf16.h>
#include <math.h>

// ---------------------------------------------------------------------------
// Model dims (fixed by the reference)
// ---------------------------------------------------------------------------
#define NB      2
#define NS      1024
#define NDIM    1024
#define NDINNER 2048
#define NDTRANK 64
#define NDSTATE 16
#define NHEADS  16
#define NKVH    4
#define NDH     64
#define MROWS   2048   // NB * NS

typedef __attribute__((ext_vector_type(16))) __bf16 v16bf;
typedef __attribute__((ext_vector_type(8)))  float  v8f;

// Tensor Data Mover availability (this toolchain: 6-arg builtin)
#if defined(__AMDGCN__) && __has_builtin(__builtin_amdgcn_tensor_load_to_lds) && \
    __has_builtin(__builtin_amdgcn_s_wait_tensorcnt)
#define USE_TDM 1
#else
#define USE_TDM 0
#endif

typedef unsigned int uint32x4 __attribute__((ext_vector_type(4)));
typedef int          int32x8  __attribute__((ext_vector_type(8)));
typedef int          int32x4  __attribute__((ext_vector_type(4)));
#if USE_TDM
typedef __attribute__((address_space(3))) unsigned short lds_us;
#endif

__device__ __forceinline__ unsigned short f2bf(float f) {
  unsigned int u = __builtin_bit_cast(unsigned int, f);
  u += 0x7FFFu + ((u >> 16) & 1u);          // round-to-nearest-even
  return (unsigned short)(u >> 16);
}

union FragBF { v16bf v; uint4 q[2]; };

// ---------------------------------------------------------------------------
// fp32 -> bf16 weight image conversion (run once per launch; weights then
// live in L2 as bf16 for all subsequent GEMMs).
// ---------------------------------------------------------------------------
__global__ __launch_bounds__(256)
void f32_to_bf16_kernel(const float* __restrict__ src,
                        unsigned short* __restrict__ dst, int n)
{
  int i = (blockIdx.x * 256 + threadIdx.x) * 4;
  if (i + 3 < n) {
    float4 v = *(const float4*)(src + i);
    ushort4 o;
    o.x = f2bf(v.x); o.y = f2bf(v.y); o.z = f2bf(v.z); o.w = f2bf(v.w);
    *(ushort4*)(dst + i) = o;
  } else {
    for (; i < n; ++i) dst[i] = f2bf(src[i]);
  }
}

// ---------------------------------------------------------------------------
// Generic WMMA GEMM:  C[m,n] = act( sum_k A[m,k]*B(n,k) + bias ) + resScale*R
//   A: fp32 [M,K] (lda).  B: bFmt==0 -> bf16 [N,K] (ldb, weights)
//                         bFmt==1 -> fp32 [K,N] (ldb, activations, omni)
//   biasMode: 0 none, 1 per-n, 2 per-m.  act: 0 none, 1 silu, 2 softplus.
//   Requires M % 128 == 0, K % 32 == 0 (always true here). N guarded.
// Tile: 128(M) x 128(N) x 32(K); 8 waves as 4x2; wave = 2x4 WMMA tiles.
// bFmt==0 + TDM: B tile staged by tensor_load_to_lds (one DMA per K-chunk,
//   issued by wave 0, double-buffered via TENSORcnt, padded rows via D# pad).
// ---------------------------------------------------------------------------
__global__ __launch_bounds__(256)
void gemm_wmma_kernel(const float* __restrict__ A, int lda,
                      const void* __restrict__ Bv, int ldb, int bFmt,
                      float* __restrict__ C, int ldc,
                      int M, int N, int K,
                      const float* __restrict__ bias, int biasMode,
                      int act,
                      const float* __restrict__ Res, float resScale)
{
  __shared__ unsigned short lsA[2][128 * 40];   // 128 x 32 bf16, stride 40
  __shared__ unsigned short lsB[2][128 * 40];   // Bt[n][k]: 128 x 32 bf16

  const int tid  = threadIdx.x;
  const int lane = tid & 31;
  const int wid  = tid >> 5;
  const int wm   = wid & 3;     // 0..3 -> 32-row strip
  const int wn   = wid >> 2;    // 0..1 -> 64-col strip
  const int hf   = lane >> 4;   // half-wave
  const int lmod = lane & 15;
  const int m0   = blockIdx.y * 128;
  const int n0   = blockIdx.x * 128;

  const float*          Bf = (const float*)Bv;
  const unsigned short* Bw = (const unsigned short*)Bv;

  const v8f vzero = {0.f, 0.f, 0.f, 0.f, 0.f, 0.f, 0.f, 0.f};
  v8f acc[2][4];
  #pragma unroll
  for (int i = 0; i < 2; ++i)
    #pragma unroll
    for (int j = 0; j < 4; ++j) acc[i][j] = vzero;

  float4 ra[4];     // A stage regs (128x32 fp32)
  uint4  rb[2];     // B stage regs, bf16 weights (fallback path)
  float4 rbf[4];    // B stage regs, fp32 [K,N]

  auto loadA = [&](int k0) {
    #pragma unroll
    for (int t = 0; t < 4; ++t) {
      int idx = tid + t * 256;
      int r = idx >> 3;
      int c = (idx & 7) << 2;
      const float* src = A + (size_t)(m0 + r) * lda + (k0 + c);
      ra[t] = *(const float4*)src;
      __builtin_prefetch(src + 32, 0, 1);          // global_prefetch_b8
    }
  };
  auto storeA = [&](int buf) {
    #pragma unroll
    for (int t = 0; t < 4; ++t) {
      int idx = tid + t * 256;
      int r = idx >> 3;
      int c = (idx & 7) << 2;
      unsigned short* dst = &lsA[buf][r * 40 + c];
      dst[0] = f2bf(ra[t].x); dst[1] = f2bf(ra[t].y);
      dst[2] = f2bf(ra[t].z); dst[3] = f2bf(ra[t].w);
    }
  };
  // fallback B path for bf16 weights
  auto loadB0 = [&](int k0) {
    #pragma unroll
    for (int t = 0; t < 2; ++t) {
      int idx = tid + t * 256;
      int r  = idx >> 2;            // 0..127 (n)
      int c8 = (idx & 3) << 3;      // 0..24  (k, 8 shorts)
      int gn = n0 + r;
      if (gn < N) rb[t] = *(const uint4*)(Bw + (size_t)gn * ldb + (k0 + c8));
      else        rb[t] = make_uint4(0u, 0u, 0u, 0u);
    }
  };
  auto storeB0 = [&](int buf) {
    #pragma unroll
    for (int t = 0; t < 2; ++t) {
      int idx = tid + t * 256;
      int r  = idx >> 2;
      int c8 = (idx & 3) << 3;
      *(uint4*)&lsB[buf][r * 40 + c8] = rb[t];
    }
  };
  // fp32 [K,N] B path (omni)
  auto loadB1 = [&](int k0) {
    #pragma unroll
    for (int t = 0; t < 4; ++t) {
      int idx = tid + t * 256;
      int kr = idx >> 5;            // 0..31 (k)
      int n4 = (idx & 31) << 2;     // 0..124 (n)
      int gn = n0 + n4;
      if (gn + 3 < N) rbf[t] = *(const float4*)(Bf + (size_t)(k0 + kr) * ldb + gn);
      else            rbf[t] = make_float4(0.f, 0.f, 0.f, 0.f);
    }
  };
  auto storeB1 = [&](int buf) {
    #pragma unroll
    for (int t = 0; t < 4; ++t) {
      int idx = tid + t * 256;
      int kr = idx >> 5;
      int n4 = (idx & 31) << 2;
      lsB[buf][(n4 + 0) * 40 + kr] = f2bf(rbf[t].x);
      lsB[buf][(n4 + 1) * 40 + kr] = f2bf(rbf[t].y);
      lsB[buf][(n4 + 2) * 40 + kr] = f2bf(rbf[t].z);
      lsB[buf][(n4 + 3) * 40 + kr] = f2bf(rbf[t].w);
    }
  };

#if USE_TDM
  // Tensor DMA: 2D D# descriptor, tile = 32(k) x 128(n) bf16 from [N,K] bf16,
  // LDS rows padded 16 DWORDs data + 4 DWORDs pad -> 40-short row stride.
  auto issueB = [&](int k0, int bufSel) {
    unsigned long long ga = (unsigned long long)(const void*)Bw
                          + (((size_t)(unsigned)n0 * (size_t)(unsigned)ldb
                              + (size_t)(unsigned)k0) << 1);
    unsigned ldsoff = (unsigned)(unsigned long long)(lds_us*)&lsB[bufSel][0];
    uint32x4 g0 = { 1u,                                   // count=1 (valid D#)
                    ldsoff,                               // lds_addr
                    (unsigned)ga,                         // global_addr lo
                    (unsigned)((ga >> 32) & 0x01FFFFFFull) | 0x80000000u }; // +type=2
    unsigned td0 = (unsigned)K, td1 = (unsigned)N;
    int32x8 g1 = {
      (int)((1u << 16)            // data_size = 2 bytes
            | (1u << 20)          // pad_enable
            | (3u << 22)          // pad_interval = 16 DWORDs
            | (3u << 25)),        // pad_amount   = 4 DWORDs
      (int)((td0 & 0xFFFFu) << 16),                       // tensor_dim0 lo
      (int)((td0 >> 16) | ((td1 & 0xFFFFu) << 16)),       // dim0 hi | dim1 lo
      (int)((td1 >> 16) | (32u << 16)),                   // dim1 hi | tile_dim0=32
      (int)128,                                           // tile_dim1=128
      (int)(unsigned)ldb,                                 // tensor_dim0_stride lo
      0, 0 };
    int32x4 gz4 = { 0, 0, 0, 0 };
    int32x8 gz8 = { 0, 0, 0, 0, 0, 0, 0, 0 };
    __builtin_amdgcn_tensor_load_to_lds(g0, g1, gz4, gz4, gz8, 0);
  };
#endif

  // prologue
  loadA(0);
#if USE_TDM
  if (bFmt == 0) { if (wid == 0) issueB(0, 0); }
  else           loadB1(0);
#else
  if (bFmt == 0) loadB0(0); else loadB1(0);
#endif

  int buf = 0;
  for (int k0 = 0; k0 < K; k0 += 32) {
    storeA(buf);
#if USE_TDM
    if (bFmt != 0) storeB1(buf);
#else
    if (bFmt == 0) storeB0(buf); else storeB1(buf);
#endif
    if (k0 + 32 < K) {
      loadA(k0 + 32);                                // in flight during WMMA
#if USE_TDM
      if (bFmt != 0) loadB1(k0 + 32);
#else
      if (bFmt == 0) loadB0(k0 + 32); else loadB1(k0 + 32);
#endif
    }
#if USE_TDM
    if (bFmt == 0 && wid == 0) __builtin_amdgcn_s_wait_tensorcnt(0);
#endif
    __syncthreads();
#if USE_TDM
    if (bFmt == 0 && wid == 0 && k0 + 32 < K) issueB(k0 + 32, buf ^ 1);
#endif

    // A frag: lane m=lmod, elem e -> k = hf*8 + (e/8)*16 + (e%8)
    // B frag: lane n=lmod, elem e -> k = hf*8 + (e/8)*16 + (e%8)
    FragBF fa[2], fb[4];
    #pragma unroll
    for (int i = 0; i < 2; ++i) {
      int row = wm * 32 + i * 16 + lmod;
      const uint4* pa = (const uint4*)(&lsA[buf][row * 40 + hf * 8]);
      fa[i].q[0] = pa[0];           // k = hf*8 + 0..7
      fa[i].q[1] = pa[2];           // k = hf*8 + 16..23
    }
    #pragma unroll
    for (int j = 0; j < 4; ++j) {
      int col = wn * 64 + j * 16 + lmod;
      const uint4* pb = (const uint4*)(&lsB[buf][col * 40 + hf * 8]);
      fb[j].q[0] = pb[0];
      fb[j].q[1] = pb[2];
    }
    #pragma unroll
    for (int i = 0; i < 2; ++i)
      #pragma unroll
      for (int j = 0; j < 4; ++j)
        acc[i][j] = __builtin_amdgcn_wmma_f32_16x16x32_bf16(
            false, fa[i].v, false, fb[j].v, (short)0, acc[i][j], false, false);
    buf ^= 1;
  }

  // ---- epilogue: C layout lane n=lmod, vgpr r -> m = hf*8 + r ----
  #pragma unroll
  for (int i = 0; i < 2; ++i) {
    #pragma unroll
    for (int j = 0; j < 4; ++j) {
      int gn = n0 + wn * 64 + j * 16 + lmod;
      if (gn < N) {
        #pragma unroll
        for (int r = 0; r < 8; ++r) {
          int gm = m0 + wm * 32 + i * 16 + hf * 8 + r;
          float v = acc[i][j][r];
          if (biasMode == 1 && bias) v += bias[gn];
          else if (biasMode == 2 && bias) v += bias[gm];
          if (act == 1)      v = v / (1.f + __expf(-v));                   // silu
          else if (act == 2) v = (v > 20.f) ? v : log1pf(__expf(v));       // softplus
          if (Res) v += resScale * Res[(size_t)gm * ldc + gn];
          C[(size_t)gm * ldc + gn] = v;
        }
      }
    }
  }
}

// ---------------------------------------------------------------------------
// RMSNorm / LayerNorm (one block per row)
// ---------------------------------------------------------------------------
__global__ __launch_bounds__(256)
void rmsnorm_kernel(const float* __restrict__ x, const float* __restrict__ g,
                    float* __restrict__ y, int D)
{
  __shared__ float red[256];
  const size_t row = blockIdx.x;
  const float* xr = x + row * D;
  float* yr = y + row * D;
  float ss = 0.f;
  for (int i = threadIdx.x; i < D; i += 256) { float v = xr[i]; ss += v * v; }
  red[threadIdx.x] = ss; __syncthreads();
  for (int off = 128; off > 0; off >>= 1) {
    if (threadIdx.x < off) red[threadIdx.x] += red[threadIdx.x + off];
    __syncthreads();
  }
  float rs = rsqrtf(red[0] / (float)D + 1e-5f);
  for (int i = threadIdx.x; i < D; i += 256) yr[i] = xr[i] * rs * g[i];
}

__global__ __launch_bounds__(256)
void layernorm_kernel(const float* __restrict__ x, const float* __restrict__ g,
                      const float* __restrict__ b, float* __restrict__ y, int D)
{
  __shared__ float s1[256];
  __shared__ float s2[256];
  const size_t row = blockIdx.x;
  const float* xr = x + row * D;
  float* yr = y + row * D;
  float sum = 0.f, sq = 0.f;
  for (int i = threadIdx.x; i < D; i += 256) { float v = xr[i]; sum += v; sq += v * v; }
  s1[threadIdx.x] = sum; s2[threadIdx.x] = sq; __syncthreads();
  for (int off = 128; off > 0; off >>= 1) {
    if (threadIdx.x < off) { s1[threadIdx.x] += s1[threadIdx.x + off];
                             s2[threadIdx.x] += s2[threadIdx.x + off]; }
    __syncthreads();
  }
  float mean = s1[0] / (float)D;
  float var  = s2[0] / (float)D - mean * mean;
  float rs = rsqrtf(var + 1e-5f);
  for (int i = threadIdx.x; i < 256 && false; ++i) {}   // no-op
  for (int i = threadIdx.x; i < D; i += 256)
    yr[i] = (xr[i] - mean) * rs * g[i] + b[i];
}

// ---------------------------------------------------------------------------
// Depthwise causal conv1d (D_CONV=4) + bias + silu.
// xz: [M,4096] (first 2048 cols = xin). out: [M,2048]
// ---------------------------------------------------------------------------
__global__ __launch_bounds__(256)
void conv_silu_kernel(const float* __restrict__ xz, const float* __restrict__ w,
                      const float* __restrict__ b, float* __restrict__ out)
{
  int idx = blockIdx.x * 256 + threadIdx.x;          // M*2048 threads
  if (idx >= MROWS * NDINNER) return;
  int m = idx / NDINNER;
  int d = idx - m * NDINNER;
  int bb = m / NS;
  int t  = m - bb * NS;
  float s = b[d];
  #pragma unroll
  for (int j = 0; j < 4; ++j) {
    int tt = t - 3 + j;
    if (tt >= 0)
      s += xz[((size_t)(bb * NS + tt)) * (2 * NDINNER) + d] * w[d * 4 + j];
  }
  out[idx] = s / (1.f + __expf(-s));
}

// ---------------------------------------------------------------------------
// Selective scan: thread per (b, d) channel, 16-state recurrence in registers.
// delta,xin,ys: [M, 2048]. dbc: [M, 96] (B at col 64, C at col 80). A_log: [2048,16]
// ---------------------------------------------------------------------------
__global__ __launch_bounds__(256)
void scan_kernel(const float* __restrict__ delta, const float* __restrict__ xin,
                 const float* __restrict__ dbc, const float* __restrict__ A_log,
                 float* __restrict__ ys)
{
  int gid = blockIdx.x * 256 + threadIdx.x;          // NB*NDINNER = 4096
  if (gid >= NB * NDINNER) return;
  int bb = gid / NDINNER;
  int d  = gid - bb * NDINNER;
  float A[NDSTATE], h[NDSTATE];
  #pragma unroll
  for (int n = 0; n < NDSTATE; ++n) { A[n] = -__expf(A_log[d * NDSTATE + n]); h[n] = 0.f; }
  for (int t = 0; t < NS; ++t) {
    size_t row = (size_t)bb * NS + t;
    float dt = delta[row * NDINNER + d];
    float xv = xin[row * NDINNER + d];
    const float* bc = dbc + row * 96;
    float y = 0.f;
    float dx = dt * xv;
    #pragma unroll
    for (int n = 0; n < NDSTATE; ++n) {
      float dA = __expf(dt * A[n]);
      h[n] = dA * h[n] + dx * bc[64 + n];
      y += h[n] * bc[80 + n];
    }
    ys[row * NDINNER + d] = y;
  }
}

// y = (ys + Dp[d]*xin) * silu(z);  z = xz[:, 2048+d].  In-place into ys.
__global__ __launch_bounds__(256)
void combine_kernel(float* __restrict__ ys, const float* __restrict__ xin,
                    const float* __restrict__ xz, const float* __restrict__ Dp)
{
  int idx = blockIdx.x * 256 + threadIdx.x;
  if (idx >= MROWS * NDINNER) return;
  int m = idx / NDINNER;
  int d = idx - m * NDINNER;
  float z = xz[(size_t)m * (2 * NDINNER) + NDINNER + d];
  float y = (ys[idx] + Dp[d] * xin[idx]) * (z / (1.f + __expf(-z)));
  ys[idx] = y;
}

// L2-normalize per-head 64-dim vectors with learned scales (qk_norm).
__global__ __launch_bounds__(256)
void qknorm_kernel(float* __restrict__ X, const float* __restrict__ scale, int heads)
{
  int gid = blockIdx.x * 256 + threadIdx.x;          // M*heads
  if (gid >= MROWS * heads) return;
  int m = gid / heads;
  int h = gid - m * heads;
  float* p = X + (size_t)m * heads * NDH + h * NDH;
  float ss = 1e-12f;
  #pragma unroll
  for (int d = 0; d < NDH; ++d) ss += p[d] * p[d];
  float rs = rsqrtf(ss);
  const float* sc = scale + h * NDH;
  #pragma unroll
  for (int d = 0; d < NDH; ++d) p[d] = p[d] * rs * sc[d];
}

// Flash-style attention, thread per query row. GQA repeat built into indexing.
__global__ __launch_bounds__(256)
void attn_kernel(const float* __restrict__ Q, const float* __restrict__ Kt,
                 const float* __restrict__ V, float* __restrict__ O)
{
  int gid = blockIdx.x * 256 + threadIdx.x;          // NB*NHEADS*NS = 32768
  int qi = gid & (NS - 1);
  int h  = (gid >> 10) & (NHEADS - 1);
  int bb = gid >> 14;
  const float* qp = Q + ((size_t)(bb * NS + qi)) * (NHEADS * NDH) + h * NDH;
  int kh = h >> 2;                                   // HEADS/KV_HEADS = 4
  const float* kbase = Kt + (size_t)bb * NS * (NKVH * NDH) + kh * NDH;
  const float* vbase = V  + (size_t)bb * NS * (NKVH * NDH) + kh * NDH;
  float q[NDH], acc[NDH];
  #pragma unroll
  for (int d = 0; d < NDH; ++d) { q[d] = qp[d]; acc[d] = 0.f; }
  float mx = -1e30f, l = 0.f;
  for (int s = 0; s < NS; ++s) {
    const float* kr = kbase + (size_t)s * (NKVH * NDH);
    float dot = 0.f;
    #pragma unroll
    for (int d = 0; d < NDH; ++d) dot += q[d] * kr[d];
    dot *= 10.0f;                                    // QK_SCALE
    float nm = fmaxf(mx, dot);
    float c  = __expf(mx - nm);
    float p  = __expf(dot - nm);
    l = l * c + p;
    const float* vr = vbase + (size_t)s * (NKVH * NDH);
    #pragma unroll
    for (int d = 0; d < NDH; ++d) acc[d] = acc[d] * c + p * vr[d];
    mx = nm;
  }
  float inv = 1.f / l;
  float* op = O + ((size_t)(bb * NS + qi)) * (NHEADS * NDH) + h * NDH;
  #pragma unroll
  for (int d = 0; d < NDH; ++d) op[d] = acc[d] * inv;
}

// ---------------------------------------------------------------------------
// Host orchestration
// ---------------------------------------------------------------------------
struct MambaP {
  const float *rms_g, *in_w, *conv_w, *conv_b, *xproj_w, *dt_w, *dt_b,
              *A_log, *Dp, *out_w, *ln_g, *ln_b;
  const unsigned short *in_wb, *xproj_wb, *dt_wb, *out_wb;   // bf16 images
};
struct SharedP {
  const float *wq, *wk, *wv, *wo, *q_scale, *k_scale, *ln_g, *ln_b,
              *ffn1_w, *ffn1_b, *ffn_ln_g, *ffn_ln_b, *ffn2_w, *ffn2_b;
  const unsigned short *wq_b, *wk_b, *wv_b, *wo_b, *ffn1_wb, *ffn2_wb;
};

extern "C" void kernel_launch(void* const* d_in, const int* in_sizes, int n_in,
                              void* d_out, int out_size, void* d_ws, size_t ws_size,
                              hipStream_t stream)
{
  auto P = [&](int i) { return (const float*)d_in[i]; };

  // --- parameter leaves (setup_inputs traversal order) ---
  const float* x0 = P(0);
  auto mkM = [&](int b) {
    MambaP p;
    p.rms_g = P(b + 0);  p.in_w  = P(b + 1);  p.conv_w = P(b + 2);
    p.conv_b = P(b + 3); p.xproj_w = P(b + 4); p.dt_w = P(b + 5);
    p.dt_b = P(b + 6);   p.A_log = P(b + 7);  p.Dp = P(b + 8);
    p.out_w = P(b + 9);  p.ln_g = P(b + 10);  p.ln_b = P(b + 11);
    p.in_wb = nullptr; p.xproj_wb = nullptr; p.dt_wb = nullptr; p.out_wb = nullptr;
    return p;
  };
  MambaP f1 = mkM(1), f2 = mkM(13);
  SharedP sp;
  sp.wq = P(25); sp.wk = P(26); sp.wv = P(27); sp.wo = P(28);
  sp.q_scale = P(29); sp.k_scale = P(30); sp.ln_g = P(31); sp.ln_b = P(32);
  sp.ffn1_w = P(33); sp.ffn1_b = P(34); sp.ffn_ln_g = P(35); sp.ffn_ln_b = P(36);
  sp.ffn2_w = P(37); sp.ffn2_b = P(38);
  const float* proj_w = P(39); const float* proj_b = P(40);
  const float* op1_w = P(41);  const float* op1_b = P(42);
  const float* op2_w = P(43);  const float* op2_b = P(44);
  const float* fin_g = P(45);  const float* fin_b = P(46);

  // --- fp32 workspace arena ---
  float* ws = (float*)d_ws;
  const size_t M = MROWS;
  float* bufNorm  = ws; ws += M * 1024;
  float* bufXZ    = ws; ws += M * 4096;
  float* bufXin   = ws; ws += M * 2048;
  float* bufDelta = ws; ws += M * 2048;
  float* bufYs    = ws; ws += M * 2048;
  float* bufDbc   = ws; ws += M * 128;      // used as [M,96]
  float* bufPA    = ws; ws += M * 1024;
  float* bufPB    = ws; ws += M * 1024;
  float* bufF1    = ws; ws += M * 1024;
  float* bufF2    = ws; ws += M * 1024;
  float* bufE     = ws; ws += M * 1024;
  float* bufQ     = ws; ws += M * 1024;
  float* bufK     = ws; ws += M * 256;
  float* bufV     = ws; ws += M * 256;
  float* bufO     = ws; ws += M * 1024;
  float* bufH2    = ws; ws += M * 4096;
  float* bufProj  = ws; ws += M * 1024;

  // --- bf16 weight-image arena (converted once per launch) ---
  unsigned short* wsb = (unsigned short*)ws;
  auto cvt = [&](const float* src, size_t n) {
    unsigned short* dst = wsb; wsb += n;
    f32_to_bf16_kernel<<<(int)((n + 1023) / 1024), 256, 0, stream>>>(src, dst, (int)n);
    return (const unsigned short*)dst;
  };
  f1.in_wb    = cvt(f1.in_w,    (size_t)4096 * 1024);
  f1.xproj_wb = cvt(f1.xproj_w, (size_t)96 * 2048);
  f1.dt_wb    = cvt(f1.dt_w,    (size_t)2048 * 64);
  f1.out_wb   = cvt(f1.out_w,   (size_t)1024 * 2048);
  f2.in_wb    = cvt(f2.in_w,    (size_t)4096 * 1024);
  f2.xproj_wb = cvt(f2.xproj_w, (size_t)96 * 2048);
  f2.dt_wb    = cvt(f2.dt_w,    (size_t)2048 * 64);
  f2.out_wb   = cvt(f2.out_w,   (size_t)1024 * 2048);
  sp.wq_b     = cvt(sp.wq,      (size_t)1024 * 1024);
  sp.wk_b     = cvt(sp.wk,      (size_t)256 * 1024);
  sp.wv_b     = cvt(sp.wv,      (size_t)256 * 1024);
  sp.wo_b     = cvt(sp.wo,      (size_t)1024 * 1024);
  sp.ffn1_wb  = cvt(sp.ffn1_w,  (size_t)4096 * 1024);
  sp.ffn2_wb  = cvt(sp.ffn2_w,  (size_t)1024 * 4096);
  const unsigned short* proj_wb = cvt(proj_w, (size_t)1024 * 1024);
  (void)ws_size; (void)in_sizes; (void)n_in; (void)out_size;

  auto gemm = [&](const float* A, int lda, const void* Bm, int ldb, int bFmt,
                  float* C, int ldc, int Mg, int Ng, int Kg,
                  const float* bias, int biasMode, int act,
                  const float* Res, float resScale) {
    dim3 g((Ng + 127) / 128, (Mg + 127) / 128);
    gemm_wmma_kernel<<<g, dim3(256), 0, stream>>>(
        A, lda, Bm, ldb, bFmt, C, ldc, Mg, Ng, Kg, bias, biasMode, act, Res, resScale);
  };

  const int EW = (int)((M * NDINNER + 255) / 256);   // elementwise grid (M*2048)

  // fractal: 6x [ x = mamba_block(rms(x)) + 2x ], then LN(ln_g, ln_b) -> dst
  auto fractal = [&](const float* input, const MambaP& mp, float* dst) {
    const float* cur = input;
    for (int it = 0; it < 6; ++it) {
      float* nxt = (it & 1) ? bufPB : bufPA;
      rmsnorm_kernel<<<MROWS, 256, 0, stream>>>(cur, mp.rms_g, bufNorm, 1024);
      gemm(bufNorm, 1024, mp.in_wb, 1024, 0, bufXZ, 4096, MROWS, 4096, 1024,
           nullptr, 0, 0, nullptr, 0.f);
      conv_silu_kernel<<<EW, 256, 0, stream>>>(bufXZ, mp.conv_w, mp.conv_b, bufXin);
      gemm(bufXin, 2048, mp.xproj_wb, 2048, 0, bufDbc, 96, MROWS, 96, 2048,
           nullptr, 0, 0, nullptr, 0.f);
      gemm(bufDbc, 96, mp.dt_wb, 64, 0, bufDelta, 2048, MROWS, 2048, 64,
           mp.dt_b, 1, 2 /*softplus*/, nullptr, 0.f);
      scan_kernel<<<(NB * NDINNER + 255) / 256, 256, 0, stream>>>(
          bufDelta, bufXin, bufDbc, mp.A_log, bufYs);
      combine_kernel<<<EW, 256, 0, stream>>>(bufYs, bufXin, bufXZ, mp.Dp);
      gemm(bufYs, 2048, mp.out_wb, 2048, 0, nxt, 1024, MROWS, 1024, 2048,
           nullptr, 0, 0, cur, 2.0f);                 // + 2x residual
      cur = nxt;
    }
    layernorm_kernel<<<MROWS, 256, 0, stream>>>(cur, mp.ln_g, mp.ln_b, dst, 1024);
  };

  // omni: E_b[t,d] = sum_s W[t,s]*skip_b[s,d] + sum_s W[t,1024+s]*xb_b[s,d] + b[t]
  auto omni = [&](const float* skip, const float* xb, const float* w, const float* bvec) {
    for (int b2 = 0; b2 < NB; ++b2) {
      const float* skipb = skip + (size_t)b2 * NS * NDIM;
      const float* xbb   = xb   + (size_t)b2 * NS * NDIM;
      float* Eb = bufE + (size_t)b2 * NS * NDIM;
      gemm(w, 2048, skipb, 1024, 1 /*fp32 [K,N]*/, Eb, 1024, NS, NDIM, NS,
           bvec, 2 /*per-m*/, 0, nullptr, 0.f);
      gemm(w + 1024, 2048, xbb, 1024, 1, Eb, 1024, NS, NDIM, NS,
           nullptr, 0, 0, Eb, 1.0f);                  // accumulate
    }
  };

  auto shared_block = [&]() {
    gemm(bufE, 1024, sp.wq_b, 1024, 0, bufQ, 1024, MROWS, 1024, 1024,
         nullptr, 0, 0, nullptr, 0.f);
    gemm(bufE, 1024, sp.wk_b, 1024, 0, bufK, 256, MROWS, 256, 1024,
         nullptr, 0, 0, nullptr, 0.f);
    gemm(bufE, 1024, sp.wv_b, 1024, 0, bufV, 256, MROWS, 256, 1024,
         nullptr, 0, 0, nullptr, 0.f);
    qknorm_kernel<<<(MROWS * NHEADS + 255) / 256, 256, 0, stream>>>(bufQ, sp.q_scale, NHEADS);
    qknorm_kernel<<<(MROWS * NKVH + 255) / 256, 256, 0, stream>>>(bufK, sp.k_scale, NKVH);
    attn_kernel<<<(NB * NHEADS * NS) / 256, 256, 0, stream>>>(bufQ, bufK, bufV, bufO);
    gemm(bufO, 1024, sp.wo_b, 1024, 0, bufProj, 1024, MROWS, 1024, 1024,
         nullptr, 0, 0, nullptr, 0.f);
    layernorm_kernel<<<MROWS, 256, 0, stream>>>(bufProj, sp.ln_g, sp.ln_b, bufE, 1024);
    gemm(bufE, 1024, sp.ffn1_wb, 1024, 0, bufXZ, 4096, MROWS, 4096, 1024,
         sp.ffn1_b, 1, 1 /*silu*/, nullptr, 0.f);
    layernorm_kernel<<<MROWS, 256, 0, stream>>>(bufXZ, sp.ffn_ln_g, sp.ffn_ln_b, bufH2, 4096);
    gemm(bufH2, 4096, sp.ffn2_wb, 4096, 0, bufProj, 1024, MROWS, 1024, 4096,
         sp.ffn2_b, 1, 0, nullptr, 0.f);
    layernorm_kernel<<<MROWS, 256, 0, stream>>>(bufProj, sp.ln_g, sp.ln_b, bufE, 1024);
  };

  // ---------------- forward ----------------
  fractal(x0, f1, bufF1);                     // first = fractal(x, frac1)
  omni(x0, bufF1, op1_w, op1_b);              // -> bufE
  shared_block();                             // -> bufE
  gemm(bufE, 1024, proj_wb, 1024, 0, bufProj, 1024, MROWS, 1024, 1024,
       proj_b, 1, 0, bufF1, 1.0f);            // proj(x) + first
  fractal(bufProj, f2, bufF2);                // second
  omni(x0, bufF2, op2_w, op2_b);
  shared_block();
  gemm(bufE, 1024, proj_wb, 1024, 0, bufProj, 1024, MROWS, 1024, 1024,
       proj_b, 1, 0, bufF2, 1.0f);            // proj(x) + second
  fractal(bufProj, f1, bufO);                 // fractal reused with frac1
  layernorm_kernel<<<MROWS, 256, 0, stream>>>(bufO, fin_g, fin_b, (float*)d_out, 1024);
}
